// CellsExtractor_16578573763169
// MI455X (gfx1250) — matrix operand order repaired and checked by reading, not compiled, not run on hardware
//
#include <hip/hip_runtime.h>

typedef __attribute__((ext_vector_type(16))) _Float16 v16h;
typedef __attribute__((ext_vector_type(8)))  _Float16 v8h;
typedef __attribute__((ext_vector_type(8)))  float    v8f;

#define BSZ   16
#define CCH   256
#define NRR   96
#define NCC   96
#define NSEQ  96
#define CP    264   // padded channel stride (halves) for Xt/Kt/Qt rows
#define NP    104   // padded position stride (halves) for Vm/Wa rows
#define LP    100   // padded float stride for logits rows

#define SMEM_BYTES (3*(96*CP*2) + 256*NP*2 + 96*LP*4 + 96*NP*2) // 263680 B < 320 KB

__device__ __forceinline__ v16h frag_contig(const _Float16* p) {
  v8h a = *(const v8h*)p;
  v8h b = *(const v8h*)(p + 8);
  return __builtin_shufflevector(a, b, 0,1,2,3,4,5,6,7,8,9,10,11,12,13,14,15);
}

// A fragment (16x32 f16): element (m,k) at base[(m0+m)*ld + k0 + k]
// lanes 0-15: halves 0..7 -> K=0..7, 8..15 -> K=16..23 ; lanes 16-31: +8
__device__ __forceinline__ v16h fragA(const _Float16* base, int ld, int m0, int k0, int lane) {
  int m  = lane & 15;
  int kg = (lane >> 4) << 3;
  const _Float16* p = base + (size_t)(m0 + m) * ld + k0 + kg;
  v8h a = *(const v8h*)p;         // K = k0+kg+0..7
  v8h b = *(const v8h*)(p + 16);  // K = k0+kg+16..23
  return __builtin_shufflevector(a, b, 0,1,2,3,4,5,6,7,8,9,10,11,12,13,14,15);
}

// B fragment (32x16 f16): element (k,n) at base[(n0+n)*ld + k0 + k]
// lanes 0-15: halves 0..7 -> K=0..7, 8..15 -> K=16..23 ; lanes 16-31: +8
__device__ __forceinline__ v16h fragB(const _Float16* base, int ld, int n0, int k0, int lane) {
  int n  = lane & 15;
  int kb = (lane >> 4) << 3;
  const _Float16* p = base + (size_t)(n0 + n) * ld + k0 + kb;
  v8h a = *(const v8h*)p;
  v8h b = *(const v8h*)(p + 16);
  return __builtin_shufflevector(a, b, 0,1,2,3,4,5,6,7,8,9,10,11,12,13,14,15);
}

__device__ __forceinline__ v8f wmma16(v16h a, v16h b, v8f c) {
  return __builtin_amdgcn_wmma_f32_16x16x32_f16(false, a, false, b, (short)0, c, false, false);
}

// One fused self-attention pass over 1536 independent sequences (one per workgroup).
__global__ void axial_sa_pass(const float* __restrict__ xsrc, float* __restrict__ xdst,
                              const _Float16* __restrict__ Wk, const float* __restrict__ Bk,
                              const _Float16* __restrict__ Wq, const float* __restrict__ Bq,
                              const _Float16* __restrict__ Wv, const float* __restrict__ Bv,
                              const int* __restrict__ num_rows, const int* __restrict__ num_cols,
                              int row_pass) {
  extern __shared__ __align__(16) char smem[];
  _Float16* Xt = (_Float16*)smem;            // [96][CP]   x, position-major f16
  _Float16* Kt = Xt + 96 * CP;               // [96][CP]   K, position-major
  _Float16* Qt = Kt + 96 * CP;               // [96][CP]   Q, position-major
  _Float16* Vm = Qt + 96 * CP;               // [256][NP]  V, channel-major
  float*    Lg = (float*)(Vm + 256 * NP);    // [96][LP]   logits f32
  _Float16* Wa = (_Float16*)(Lg + 96 * LP);  // [96][NP]   softmax weights f16

  const int tid  = threadIdx.x;
  const int lane = tid & 31;
  const int wave = tid >> 5;

  const int b = blockIdx.x / 96;
  const int s = blockIdx.x % 96;  // row idx (col pass) or col idx (row pass)

  const size_t base  = (size_t)b * (CCH * NRR * NCC) + (row_pass ? s : s * NCC);
  const int strideN  = row_pass ? NCC : 1;
  const int cstride  = NRR * NCC;

  int Lvalid;
  if (row_pass) Lvalid = (s < num_cols[b]) ? num_rows[b] : 0;
  else          Lvalid = (s < num_rows[b]) ? num_cols[b] : 0;

  // ---- Stage 1: load sequence x[C, N] -> Xt[n][c] as f16 ----
  for (int i = tid; i < CCH * NSEQ; i += blockDim.x) {
    int c = i / NSEQ;
    int n = i - c * NSEQ;
    Xt[n * CP + c] = (_Float16)xsrc[base + (size_t)c * cstride + (size_t)n * strideN];
  }
  __syncthreads();

  // ---- Stage 2: K/Q/V = W @ x + b  (three 256x96x256 GEMMs) ----
  // Job = (projection, channel-tile); weight A-fragments for the whole K dim
  // are cached in registers (64 VGPRs) and reused across all 6 N tiles.
  // Occupancy is LDS-bound (1 WG/WGP), so the VGPR budget is effectively free.
  for (int pm = wave; pm < 3 * 16; pm += 8) {
    int p  = pm / 16;        // 0=K 1=Q 2=V
    int mt = pm - p * 16;    // channel tile
    const _Float16* W  = (p == 0) ? Wk : (p == 1) ? Wq : Wv;
    const float*    Bb = (p == 0) ? Bk : (p == 1) ? Bq : Bv;

    v16h aF[8];
#pragma unroll
    for (int k = 0; k < 8; ++k) aF[k] = fragA(W, 256, mt * 16, k * 32, lane);

    const int mrow = mt * 16 + ((lane >> 4) << 3);  // output channel base for this lane
    float bv[8];
#pragma unroll
    for (int v = 0; v < 8; ++v) bv[v] = Bb[mrow + v];

    for (int nt = 0; nt < 6; ++nt) {
      v8f acc = {};
#pragma unroll
      for (int k = 0; k < 8; ++k)
        acc = wmma16(aF[k], fragB(Xt, CP, nt * 16, k * 32, lane), acc);

      int n = nt * 16 + (lane & 15);  // position
      if (p < 2) {
        _Float16* dst = ((p == 0) ? Kt : Qt) + n * CP + mrow;
        v8h hh;
#pragma unroll
        for (int v = 0; v < 8; ++v) hh[v] = (_Float16)(acc[v] + bv[v]);
        *(v8h*)dst = hh;
      } else {
#pragma unroll
        for (int v = 0; v < 8; ++v)
          Vm[(mrow + v) * NP + n] = (_Float16)(acc[v] + bv[v]);
      }
    }
  }
  __syncthreads();

  const float scale = 0.125f;  // 1/sqrt(64)

  for (int h = 0; h < 4; ++h) {
    // ---- Stage 3a: logits[n][m] = scale * sum_d K[d,n] Q[d,m] ----
    for (int job = wave; job < 36; job += 8) {
      int nt = job / 6;
      int mt = job - nt * 6;
      v8f acc = {};
#pragma unroll
      for (int kk = 0; kk < 64; kk += 32) {
        v16h a  = fragA(Kt, CP, nt * 16, h * 64 + kk, lane);
        v16h bf = fragB(Qt, CP, mt * 16, h * 64 + kk, lane);
        acc = wmma16(a, bf, acc);
      }
      int nrow = nt * 16 + ((lane >> 4) << 3);
      int mcol = mt * 16 + (lane & 15);
#pragma unroll
      for (int v = 0; v < 8; ++v) Lg[(nrow + v) * LP + mcol] = acc[v] * scale;
    }
    __syncthreads();

    // ---- Stage 3b: masked softmax over m (one row per thread) ----
    if (tid < NSEQ) {
      float*    row  = Lg + tid * LP;
      _Float16* wrow = Wa + tid * NP;
      float mx = -3.0e38f;
      for (int m = 0; m < Lvalid; ++m) mx = fmaxf(mx, row[m]);
      float sum = 0.0f;
      for (int m = 0; m < Lvalid; ++m) { float e = __expf(row[m] - mx); row[m] = e; sum += e; }
      float inv = (Lvalid > 0) ? (1.0f / sum) : 0.0f;
      for (int m = 0; m < NSEQ; ++m)
        wrow[m] = (_Float16)((m < Lvalid) ? row[m] * inv : 0.0f);
    }
    __syncthreads();

    // ---- Stage 3c: out[c][n] = sum_m V[c][m] * Wa[n][m]; +residual; store ----
    for (int job = wave; job < 24; job += 8) {
      int dt = job / 6;
      int nt = job - dt * 6;
      v8f acc = {};
#pragma unroll
      for (int k0 = 0; k0 < 96; k0 += 32) {
        v16h a  = fragA(Vm, NP, h * 64 + dt * 16, k0, lane);
        v16h bf = fragB(Wa, NP, nt * 16, k0, lane);
        acc = wmma16(a, bf, acc);
      }
      int crow = h * 64 + dt * 16 + ((lane >> 4) << 3);
      int n    = nt * 16 + (lane & 15);
#pragma unroll
      for (int v = 0; v < 8; ++v) {
        size_t g = base + (size_t)(crow + v) * cstride + (size_t)n * strideN;
        xdst[g] = acc[v] + xsrc[g];
      }
    }
    __syncthreads();
  }
}

// Convert all 6 weight matrices (2 heads each) f32 -> f16 into workspace.
__global__ void convert_weights(const float* __restrict__ w0, const float* __restrict__ w1,
                                const float* __restrict__ w2, const float* __restrict__ w3,
                                const float* __restrict__ w4, const float* __restrict__ w5,
                                _Float16* __restrict__ out) {
  int idx  = blockIdx.x * 256 + threadIdx.x;   // 12*65536 total
  int name = idx / 131072;
  int rem  = idx - name * 131072;
  const float* w = (name == 0) ? w0 : (name == 1) ? w1 : (name == 2) ? w2
                 : (name == 3) ? w3 : (name == 4) ? w4 : w5;
  out[(size_t)name * 131072 + rem] = (_Float16)w[rem];
}

__global__ void write_masks(const int* __restrict__ nr, const int* __restrict__ nc,
                            float* __restrict__ m) {
  int idx = blockIdx.x * 256 + threadIdx.x;    // 16*96*96
  int b   = idx / (NRR * NCC);
  int rem = idx - b * (NRR * NCC);
  int r   = rem / NCC;
  int c   = rem - r * NCC;
  m[idx] = (r < nr[b] && c < nc[b]) ? 1.0f : 0.0f;
}

extern "C" void kernel_launch(void* const* d_in, const int* in_sizes, int n_in,
                              void* d_out, int out_size, void* d_ws, size_t ws_size,
                              hipStream_t stream) {
  (void)in_sizes; (void)n_in; (void)out_size; (void)ws_size;

  const float* feats = (const float*)d_in[0];
  const int*   nrows = (const int*)d_in[1];
  const int*   ncols = (const int*)d_in[2];
  // name order: 0=col_k 1=col_q 2=col_v 3=row_k 4=row_q 5=row_v
  const float* W32[6];
  const float* B32[6];
  for (int i = 0; i < 6; ++i) {
    W32[i] = (const float*)d_in[3 + 2 * i];
    B32[i] = (const float*)d_in[4 + 2 * i];
  }

  const size_t XELEMS = (size_t)BSZ * CCH * NRR * NCC;
  float*    xbuf = (float*)d_ws;
  _Float16* wh   = (_Float16*)((char*)d_ws + XELEMS * sizeof(float));

  hipFuncSetAttribute((const void*)axial_sa_pass,
                      hipFuncAttributeMaxDynamicSharedMemorySize, SMEM_BYTES);

  convert_weights<<<(12 * 65536) / 256, 256, 0, stream>>>(W32[0], W32[1], W32[2],
                                                          W32[3], W32[4], W32[5], wh);
  write_masks<<<(BSZ * NRR * NCC) / 256, 256, 0, stream>>>(nrows, ncols,
                                                           (float*)d_out + XELEMS);

  float* xout = (float*)d_out;
  for (int head = 0; head < 2; ++head) {
    const float* src = (head == 0) ? feats : xbuf;
    // column attention (sequences along nc)
    axial_sa_pass<<<BSZ * NRR, 256, SMEM_BYTES, stream>>>(
        src, xbuf,
        wh + (0 * 2 + head) * 65536, B32[0] + head * 256,
        wh + (1 * 2 + head) * 65536, B32[1] + head * 256,
        wh + (2 * 2 + head) * 65536, B32[2] + head * 256,
        nrows, ncols, 0);
    // row attention (sequences along nr)
    float* dst = (head == 1) ? xout : xbuf;
    axial_sa_pass<<<BSZ * NCC, 256, SMEM_BYTES, stream>>>(
        xbuf, dst,
        wh + (3 * 2 + head) * 65536, B32[3] + head * 256,
        wh + (4 * 2 + head) * 65536, B32[4] + head * 256,
        wh + (5 * 2 + head) * 65536, B32[5] + head * 256,
        nrows, ncols, 1);
  }
}
